// ColorRouteGNN_78030965834144
// MI455X (gfx1250) — matrix operand order repaired and checked by reading, not compiled, not run on hardware
//
#include <hip/hip_runtime.h>
#include <hip/hip_bf16.h>

#define NN 50000
#define NE 800000
#define E2 (NE + NN)   // edges + self loops

typedef __attribute__((ext_vector_type(16))) _Float16 v16h;
typedef __attribute__((ext_vector_type(8)))  _Float16 v8h;
typedef __attribute__((ext_vector_type(4)))  _Float16 v4h;
typedef __attribute__((ext_vector_type(8)))  float    v8f;

union frag16 { v16h v; v8h h[2]; };

// float atomic max via ordered int/uint trick (init to -FLT_MAX)
__device__ __forceinline__ void atomicMaxF(float* addr, float val) {
  if (val >= 0.0f) atomicMax((int*)addr, __float_as_int(val));
  else             atomicMin((unsigned int*)addr, (unsigned int)__float_as_int(val));
}

// ---------------- one-off: W[k,N] (f32) -> Wt[n,K] (f16) -------------------------
__global__ void transpose_w16(const float* __restrict__ W, _Float16* __restrict__ Wt,
                              int K, int Ncols) {
  int i = blockIdx.x * blockDim.x + threadIdx.x;
  if (i >= K * Ncols) return;
  int k = i / Ncols, n = i % Ncols;
  Wt[n * K + k] = (_Float16)W[i];
}

// ---------------- node GEMM: xw[N,256] = in[N,64] @ W[64,256] (f16 WMMA) ------------
// 64 rows/block (row base clamped so the tail block overlaps -> no guards anywhere),
// 8 waves x 32 cols, B-fragments hoisted across 4 row-tiles. Wt is f16 [256][64].
__global__ __launch_bounds__(256) void gemm_node(const float* __restrict__ in,
                                                 const _Float16* __restrict__ Wt,
                                                 float* __restrict__ xw) {
  __shared__ _Float16 sWt[256 * 72];   // sWt[n*72 + k], 144B row stride (16B mult)
  __shared__ _Float16 sX[64 * 72];     // sX[m*72 + k]
  const int tid = threadIdx.x;
  int r0 = blockIdx.x * 64;
  if (r0 + 64 > NN) r0 = NN - 64;      // overlap tail: rewrites identical values
  // stage Wt: 2048 16B chunks, vectorized both sides
  for (int i = tid; i < 2048; i += 256) {
    int n = i >> 3, kc = (i & 7) << 3;
    *(v8h*)&sWt[n * 72 + kc] = ((const v8h*)Wt)[i];
  }
  // stage X tile (64 x 64)
  for (int i = tid; i < 1024; i += 256) {        // 64 rows x 16 float4
    int m = i >> 4, kc = (i & 15) << 2;
    float4 v = *(const float4*)(in + (r0 + m) * 64 + kc);
    v4h p; p[0] = (_Float16)v.x; p[1] = (_Float16)v.y;
           p[2] = (_Float16)v.z; p[3] = (_Float16)v.w;
    *(v4h*)&sX[m * 72 + kc] = p;
  }
  __syncthreads();
  const int lane = tid & 31;
  const int wv   = tid >> 5;
  const int ln   = lane & 15;
  const int hi   = lane >> 4;
  const int n0   = wv * 32;
  v8f acc[4][2];
  #pragma unroll
  for (int rt = 0; rt < 4; ++rt) { acc[rt][0] = (v8f){}; acc[rt][1] = (v8f){}; }
  #pragma unroll
  for (int kb = 0; kb < 64; kb += 32) {
    // B fragments (shared across the 4 row-tiles)
    frag16 fb0, fb1;
    const _Float16* bp0 = &sWt[(n0 + ln) * 72 + kb + (hi << 3)];
    fb0.h[0] = *(const v8h*)bp0;  fb0.h[1] = *(const v8h*)(bp0 + 16);
    const _Float16* bp1 = &sWt[(n0 + 16 + ln) * 72 + kb + (hi << 3)];
    fb1.h[0] = *(const v8h*)bp1;  fb1.h[1] = *(const v8h*)(bp1 + 16);
    #pragma unroll
    for (int rt = 0; rt < 4; ++rt) {
      frag16 fa;
      const _Float16* ap = &sX[(rt * 16 + ln) * 72 + kb + (hi << 3)];
      fa.h[0] = *(const v8h*)ap;
      fa.h[1] = *(const v8h*)(ap + 16);
      acc[rt][0] = __builtin_amdgcn_wmma_f32_16x16x32_f16(false, fa.v, false, fb0.v, (short)0, acc[rt][0], false, false);
      acc[rt][1] = __builtin_amdgcn_wmma_f32_16x16x32_f16(false, fa.v, false, fb1.v, (short)0, acc[rt][1], false, false);
    }
  }
  #pragma unroll
  for (int rt = 0; rt < 4; ++rt) {
    int row = r0 + rt * 16 + (hi << 3);
    #pragma unroll
    for (int i = 0; i < 8; ++i) {
      xw[(row + i) * 256 + n0 + ln]      = acc[rt][0][i];
      xw[(row + i) * 256 + n0 + 16 + ln] = acc[rt][1][i];
    }
  }
}

// ---------------- per-layer init ------------------------------------------------
__global__ void init_layer(float* __restrict__ msg, float* __restrict__ amax,
                           float* __restrict__ den) {
  int i = blockIdx.x * blockDim.x + threadIdx.x;
  if (i < NN * 256) msg[i] = 0.0f;
  if (i < NN * 4) { amax[i] = -3.402823466e38f; den[i] = 0.0f; }
}

// ---------------- attention scores s_src/s_dst [N,4] ----------------------------
__global__ void node_scores(const float* __restrict__ xw, const float* __restrict__ a_src,
                            const float* __restrict__ a_dst,
                            float* __restrict__ ssrc, float* __restrict__ sdst) {
  int i = blockIdx.x * blockDim.x + threadIdx.x;   // (n, h)
  if (i >= NN * 4) return;
  int n = i >> 2, h = i & 3;
  const float* xr = xw + n * 256 + h * 64;
  const float* as = a_src + h * 64;
  const float* ad = a_dst + h * 64;
  float s0 = 0.f, s1 = 0.f;
  #pragma unroll 8
  for (int c = 0; c < 64; ++c) { float v = xr[c]; s0 += v * as[c]; s1 += v * ad[c]; }
  ssrc[i] = s0; sdst[i] = s1;
}

__device__ __forceinline__ void edge_nodes(const long long* ei, int e, int& s, int& d) {
  if (e < NE) { s = (int)ei[e]; d = (int)ei[NE + e]; }
  else        { s = d = e - NE; }
}

// ---------------- segment max over dst ------------------------------------------
__global__ void edge_max_k(const long long* __restrict__ ei, const float* __restrict__ ssrc,
                           const float* __restrict__ sdst, float* __restrict__ amax) {
  int e = blockIdx.x * blockDim.x + threadIdx.x;
  if (e >= E2) return;
  int s, d; edge_nodes(ei, e, s, d);
  #pragma unroll
  for (int h = 0; h < 4; ++h) {
    float a = ssrc[s * 4 + h] + sdst[d * 4 + h];
    a = (a > 0.f) ? a : 0.2f * a;
    atomicMaxF(&amax[d * 4 + h], a);
  }
}

// ---------------- exp + segment sum ---------------------------------------------
__global__ void edge_exp_k(const long long* __restrict__ ei, const float* __restrict__ ssrc,
                           const float* __restrict__ sdst, const float* __restrict__ amax,
                           float* __restrict__ ealpha, float* __restrict__ den) {
  int e = blockIdx.x * blockDim.x + threadIdx.x;
  if (e >= E2) return;
  int s, d; edge_nodes(ei, e, s, d);
  #pragma unroll
  for (int h = 0; h < 4; ++h) {
    float a = ssrc[s * 4 + h] + sdst[d * 4 + h];
    a = (a > 0.f) ? a : 0.2f * a;
    float w = __expf(a - amax[d * 4 + h]);
    ealpha[e * 4 + h] = w;
    atomicAdd(&den[d * 4 + h], w);
  }
}

// ---------------- message aggregation (one wave per edge) ------------------------
__global__ void edge_agg_k(const long long* __restrict__ ei, const float* __restrict__ xw,
                           const float* __restrict__ ealpha, const float* __restrict__ den,
                           float* __restrict__ msg) {
  int gid  = blockIdx.x * blockDim.x + threadIdx.x;
  int e    = gid >> 5;
  int lane = gid & 31;
  if (e >= E2) return;
  int s, d; edge_nodes(ei, e, s, d);
  float al[4];
  #pragma unroll
  for (int h = 0; h < 4; ++h)
    al[h] = ealpha[e * 4 + h] / (den[d * 4 + h] + 1e-16f);
  #pragma unroll
  for (int q = 0; q < 8; ++q) {
    int idx = q * 32 + lane;
    atomicAdd(&msg[d * 256 + idx], xw[s * 256 + idx] * al[q >> 1]);
  }
}

// ---------------- head mean + bias + relu ---------------------------------------
__global__ void node_finish(const float* __restrict__ msg, const float* __restrict__ b,
                            float* __restrict__ hout) {
  int i = blockIdx.x * blockDim.x + threadIdx.x;
  if (i >= NN * 64) return;
  int n = i >> 6, c = i & 63;
  const float* mr = msg + n * 256;
  float v = 0.25f * (mr[c] + mr[64 + c] + mr[128 + c] + mr[192 + c]) + b[c];
  hout[i] = v > 0.f ? v : 0.f;
}

// ---------------- color head: h2@Wc + bc + x[:, :16] -----------------------------
__global__ void color_head(const float* __restrict__ h2, const float* __restrict__ Wc,
                           const float* __restrict__ bc, const float* __restrict__ x,
                           float* __restrict__ out) {
  int i = blockIdx.x * blockDim.x + threadIdx.x;
  if (i >= NN * 16) return;
  int n = i >> 4, j = i & 15;
  float s = bc[j] + x[n * 64 + j];
  const float* hr = h2 + n * 64;
  #pragma unroll 8
  for (int c = 0; c < 64; ++c) s += hr[c] * Wc[c * 16 + j];
  out[i] = s;
}

// ---------------- route MLP: relu([h2[s]||h2[d]] @ We1 + be1) @ We2 + be2 --------
// One wave per 16 edges; 16 WMMAs per wave. We1t pre-transposed f16 [64][128].
__global__ __launch_bounds__(256) void route_mlp(const float* __restrict__ h2,
                                                 const long long* __restrict__ ei,
                                                 const _Float16* __restrict__ We1t,
                                                 const float* __restrict__ be1,
                                                 const float* __restrict__ We2,
                                                 const float* __restrict__ be2,
                                                 float* __restrict__ out) {
  __shared__ _Float16 sW1t[64 * 136];       // sW1t[n*136 + k], 272B row stride
  __shared__ _Float16 sEF[8][16 * 136];     // ef rows, 272B stride
  __shared__ float    sB1[64];
  __shared__ float    sW2v[64];
  const int tid = threadIdx.x;
  for (int i = tid; i < 1024; i += 256) {   // 8192 halves as 16B chunks
    int n = i >> 4, kc = (i & 15) << 3;
    *(v8h*)&sW1t[n * 136 + kc] = ((const v8h*)We1t)[i];
  }
  if (tid < 64) { sB1[tid] = be1[tid]; sW2v[tid] = We2[tid]; }
  const int lane = tid & 31;
  const int wv   = tid >> 5;
  const int e0   = (blockIdx.x * 8 + wv) * 16;
  {  // stage ef tile: lane half 0 loads h2[src], half 1 loads h2[dst]
    int m = lane & 15, half = lane >> 4;
    long long node = half ? ei[NE + e0 + m] : ei[e0 + m];
    const float* hr = h2 + (int)node * 64;
    _Float16* drow = &sEF[wv][m * 136 + half * 64];
    #pragma unroll
    for (int c = 0; c < 64; c += 8) {
      float4 a = *(const float4*)(hr + c);
      float4 b = *(const float4*)(hr + c + 4);
      v8h p;
      p[0] = (_Float16)a.x; p[1] = (_Float16)a.y; p[2] = (_Float16)a.z; p[3] = (_Float16)a.w;
      p[4] = (_Float16)b.x; p[5] = (_Float16)b.y; p[6] = (_Float16)b.z; p[7] = (_Float16)b.w;
      *(v8h*)&drow[c] = p;
    }
  }
  __syncthreads();
  const int ln = lane & 15;
  const int hi = lane >> 4;
  v8f acc[4] = { {}, {}, {}, {} };
  #pragma unroll
  for (int kb = 0; kb < 128; kb += 32) {
    frag16 fa;
    const _Float16* ap = &sEF[wv][ln * 136 + kb + (hi << 3)];
    fa.h[0] = *(const v8h*)ap;          // K = kb + 8*hi + 0..7
    fa.h[1] = *(const v8h*)(ap + 16);   // K = kb + 16 + 8*hi + 0..7
    #pragma unroll
    for (int t = 0; t < 4; ++t) {
      frag16 fb;
      const _Float16* bp = &sW1t[(t * 16 + ln) * 136 + kb + (hi << 3)];
      fb.h[0] = *(const v8h*)bp;        // K = kb + 8*hi + 0..7
      fb.h[1] = *(const v8h*)(bp + 16); // K = kb + 16 + 8*hi + 0..7
      acc[t] = __builtin_amdgcn_wmma_f32_16x16x32_f16(false, fa.v, false, fb.v, (short)0, acc[t], false, false);
    }
  }
  // epilogue: relu(+be1), scale by We2[n], reduce across the 16-lane column group
  float part[8];
  #pragma unroll
  for (int i = 0; i < 8; ++i) part[i] = 0.f;
  #pragma unroll
  for (int t = 0; t < 4; ++t) {
    int n = t * 16 + ln;
    float bn = sB1[n], wn = sW2v[n];
    #pragma unroll
    for (int i = 0; i < 8; ++i) {
      float y = acc[t][i] + bn;
      y = y > 0.f ? y : 0.f;
      part[i] += y * wn;
    }
  }
  #pragma unroll
  for (int i = 0; i < 8; ++i) {
    #pragma unroll
    for (int off = 1; off < 16; off <<= 1)
      part[i] += __shfl_xor(part[i], off, 32);
  }
  if (ln == 0) {   // lane 0 -> rows 0..7, lane 16 -> rows 8..15
    float bb = be2[0];
    #pragma unroll
    for (int i = 0; i < 8; ++i)
      out[e0 + (hi << 3) + i] = part[i] + bb;
  }
}

extern "C" void kernel_launch(void* const* d_in, const int* in_sizes, int n_in,
                              void* d_out, int out_size, void* d_ws, size_t ws_size,
                              hipStream_t stream) {
  const float*     x   = (const float*)d_in[0];
  const long long* ei  = (const long long*)d_in[1];   // int64 edge_index (2, E)
  const float*     W1  = (const float*)d_in[2];
  const float*     as1 = (const float*)d_in[3];
  const float*     ad1 = (const float*)d_in[4];
  const float*     b1  = (const float*)d_in[5];
  const float*     W2  = (const float*)d_in[6];
  const float*     as2 = (const float*)d_in[7];
  const float*     ad2 = (const float*)d_in[8];
  const float*     b2  = (const float*)d_in[9];
  const float*     Wc  = (const float*)d_in[10];
  const float*     bc  = (const float*)d_in[11];
  const float*     We1 = (const float*)d_in[12];
  const float*     be1 = (const float*)d_in[13];
  const float*     We2 = (const float*)d_in[14];
  const float*     be2 = (const float*)d_in[15];

  float* out_color = (float*)d_out;                    // [N,16]
  float* out_route = (float*)d_out + (size_t)NN * 16;  // [E]

  float* ws = (float*)d_ws;
  size_t off = 0;
  float* xw   = ws + off; off += (size_t)NN * 256;
  float* msg  = ws + off; off += (size_t)NN * 256;
  float* ssrc = ws + off; off += (size_t)NN * 4;
  float* sdst = ws + off; off += (size_t)NN * 4;
  float* amax = ws + off; off += (size_t)NN * 4;
  float* den  = ws + off; off += (size_t)NN * 4;
  float* ealp = ws + off; off += (size_t)E2 * 4;
  float* h1   = ws + off; off += (size_t)NN * 64;
  float* h2   = ws + off; off += (size_t)NN * 64;
  // f16 transposed weights (16B-aligned: all float offsets are multiples of 4)
  _Float16* W1t  = (_Float16*)(ws + off); off += (size_t)(64 * 256) / 2;
  _Float16* W2t  = (_Float16*)(ws + off); off += (size_t)(64 * 256) / 2;
  _Float16* We1t = (_Float16*)(ws + off); off += (size_t)(128 * 64) / 2;

  // one-off weight transposes to f16
  transpose_w16<<<(64 * 256 + 255) / 256, 256, 0, stream>>>(W1, W1t, 64, 256);
  transpose_w16<<<(64 * 256 + 255) / 256, 256, 0, stream>>>(W2, W2t, 64, 256);
  transpose_w16<<<(128 * 64 + 255) / 256, 256, 0, stream>>>(We1, We1t, 128, 64);

  auto layer = [&](const float* hin, const _Float16* Wt, const float* as, const float* ad,
                   const float* b, float* hout) {
    gemm_node  <<<(NN + 63) / 64, 256, 0, stream>>>(hin, Wt, xw);
    init_layer <<<(NN * 256 + 255) / 256, 256, 0, stream>>>(msg, amax, den);
    node_scores<<<(NN * 4 + 255) / 256, 256, 0, stream>>>(xw, as, ad, ssrc, sdst);
    edge_max_k <<<(E2 + 255) / 256, 256, 0, stream>>>(ei, ssrc, sdst, amax);
    edge_exp_k <<<(E2 + 255) / 256, 256, 0, stream>>>(ei, ssrc, sdst, amax, ealp, den);
    edge_agg_k <<<((size_t)E2 * 32 + 255) / 256, 256, 0, stream>>>(ei, xw, ealp, den, msg);
    node_finish<<<(NN * 64 + 255) / 256, 256, 0, stream>>>(msg, b, hout);
  };

  layer(x, W1t, as1, ad1, b1, h1);
  layer(h1, W2t, as2, ad2, b2, h2);
  color_head<<<(NN * 16 + 255) / 256, 256, 0, stream>>>(h2, Wc, bc, x, out_color);
  route_mlp <<<NE / 128, 256, 0, stream>>>(h2, ei, We1t, be1, We2, be2, out_route);
}
